// Attention_12343736009384
// MI455X (gfx1250) — matrix operand order repaired and checked by reading, not compiled, not run on hardware
//
#include <hip/hip_runtime.h>
#include <hip/hip_bf16.h>

// Problem constants (match the reference)
#define B_    2
#define S_    2048
#define DIM_  1024
#define H_    16
#define HD_   64

typedef __attribute__((ext_vector_type(16))) _Float16 v16h;
typedef __attribute__((ext_vector_type(8)))  _Float16 v8h;
typedef __attribute__((ext_vector_type(8)))  float    v8f;
typedef __attribute__((ext_vector_type(4)))  unsigned int v4u;
typedef __attribute__((ext_vector_type(8)))  int      v8i;
typedef __attribute__((ext_vector_type(4)))  int      v4i;

// Toolchain arity probe for the TDM builtin:
//  - amdgpu-toolchain (therock headers) ships the TDM header and uses 6 args
//  - ROCm 7.2 uses 5 args
#if __has_include(<hip/amd_detail/amd_gfx1250_TDM.h>)
#define TDM_6ARG 1
#endif

static __device__ __forceinline__ v8h ld8(const _Float16* p) {
    return *reinterpret_cast<const v8h*>(p);
}

static __device__ __forceinline__ v16h cat16(v8h a, v8h b) {
    v16h r;
#pragma unroll
    for (int i = 0; i < 8; ++i) { r[i] = a[i]; r[i + 8] = b[i]; }
    return r;
}

static __device__ __forceinline__ v8f wmma_f16(v16h a, v16h b, v8f c) {
    // D = A(16x32 f16) * B(32x16 f16) + C(16x16 f32)
    return __builtin_amdgcn_wmma_f32_16x16x32_f16(
        /*neg_a=*/false, a, /*neg_b=*/false, b,
        /*c_mod=*/(short)0, c, /*reuse_a=*/false, /*reuse_b=*/false);
}

// ---------------------------------------------------------------------------
// Tensor Data Mover: 2D tile load Global -> LDS.
// D# bitfields per CDNA5 ISA:
//   group0: [1:0]=count, [63:32]=lds_addr, [120:64]=global_addr, [127:126]=2
//   group1: [17:16]=data_size(1 -> 2B), [79:48]=tensor_dim0, [111:80]=tensor_dim1,
//           [127:112]=tile_dim0, [143:128]=tile_dim1, [207:160]=tensor_dim0_stride
// Dims/strides in data_size units (f16 elements). Tiles fully in-bounds here.
// ---------------------------------------------------------------------------
static __device__ __forceinline__ void tdm_load_2d(
    unsigned lds_byte_addr, const _Float16* gptr,
    unsigned tensor_d0, unsigned tensor_d1, unsigned stride_d0,
    unsigned tile_d0, unsigned tile_d1)
{
    unsigned long long ga = (unsigned long long)(uintptr_t)gptr;
    v4u g0;
    g0[0] = 1u;                                            // count=1 (valid D#)
    g0[1] = lds_byte_addr;                                 // LDS dest (bytes)
    g0[2] = (unsigned)(ga & 0xFFFFFFFFu);                  // global_addr lo
    g0[3] = (unsigned)((ga >> 32) & 0x01FFFFFFu)           // global_addr hi
          | (2u << 30);                                    // type = 2 ("image")
    v8i g1;
    g1[0] = (int)(1u << 16);                               // data_size = 2 bytes
    g1[1] = (int)((tensor_d0 & 0xFFFFu) << 16);            // tensor_dim0 lo16
    g1[2] = (int)((tensor_d0 >> 16) | ((tensor_d1 & 0xFFFFu) << 16));
    g1[3] = (int)((tensor_d1 >> 16) | (tile_d0 << 16));    // tile_dim0
    g1[4] = (int)tile_d1;                                  // tile_dim1 (dim2=0)
    g1[5] = (int)stride_d0;                                // dim0 stride lo32
    g1[6] = 0;                                             // stride hi / dim1 stride
    g1[7] = 0;
    v4i gz = {0, 0, 0, 0};
#if defined(TDM_6ARG)
    v8i gz8 = {0, 0, 0, 0, 0, 0, 0, 0};
    __builtin_amdgcn_tensor_load_to_lds(g0, g1, gz, gz, gz8, 0);
#else
    __builtin_amdgcn_tensor_load_to_lds(g0, g1, gz, gz, 0);
#endif
}

static __device__ __forceinline__ unsigned lds_offset(const void* p) {
    // Generic pointers to LDS carry the wave-relative LDS offset in addr[31:0]
    // (flat aperture mapping).
    return (unsigned)(uintptr_t)p;
}

// ---------------------------------------------------------------------------
// Prep: f32 -> f16 conversion of activations
// ---------------------------------------------------------------------------
__global__ void cvt_f32_to_f16_kernel(const float* __restrict__ src,
                                      _Float16* __restrict__ dst, int n) {
    int i = blockIdx.x * blockDim.x + threadIdx.x;
    if (i < n) dst[i] = (_Float16)src[i];
}

// Prep: transpose 1024x1024 f32 [k][n] -> f16 [n][k] so B-fragments are
// contiguous along K for every output column n.
__global__ void transpose_w_kernel(const float* __restrict__ src,
                                   _Float16* __restrict__ dst) {
    int i = blockIdx.x * blockDim.x + threadIdx.x;   // i = n*1024 + k
    int n = i >> 10;
    int k = i & 1023;
    dst[i] = (_Float16)src[k * DIM_ + n];
}

// ---------------------------------------------------------------------------
// QKV projection GEMM: [4096 x 1024] @ [1024 x 1024] per matrix.
// One wave computes a 32x64 tile: 8 WMMA accumulators, 2 A fragments shared
// across 4 B fragments (1.5 b128 loads per WMMA). All fragments are made
// live before the WMMA burst so loads pipeline across iterations.
// q,k stored [b][h][s][d]; v stored transposed [b][h][d][s] for P*V.
// ---------------------------------------------------------------------------
__global__ __launch_bounds__(32)
void qkv_gemm_kernel(const _Float16* __restrict__ xh,
                     const _Float16* __restrict__ wqt,
                     const _Float16* __restrict__ wkt,
                     const _Float16* __restrict__ wvt,
                     const float* __restrict__ bq,
                     const float* __restrict__ bk,
                     const float* __restrict__ bv,
                     _Float16* __restrict__ qh,
                     _Float16* __restrict__ kh,
                     _Float16* __restrict__ vh) {
    const int lane = threadIdx.x & 31;
    const int col  = lane & 15;
    const int hi   = lane >> 4;
    const int n0    = blockIdx.x * 64;  // 16 strips of 64 columns
    const int m0    = blockIdx.y * 32;  // 128 strips of 32 rows
    const int which = blockIdx.z;       // 0=q 1=k 2=v

    const _Float16* wt   = (which == 0) ? wqt : ((which == 1) ? wkt : wvt);
    const float*    bias = (which == 0) ? bq  : ((which == 1) ? bk  : bv);

    const _Float16* ap0 = xh + (size_t)(m0 + col) * DIM_;       // A rows 0..15
    const _Float16* ap1 = ap0 + (size_t)16 * DIM_;              // A rows 16..31
    const _Float16* bp0 = wt + (size_t)(n0 +  0 + col) * DIM_;
    const _Float16* bp1 = wt + (size_t)(n0 + 16 + col) * DIM_;
    const _Float16* bp2 = wt + (size_t)(n0 + 32 + col) * DIM_;
    const _Float16* bp3 = wt + (size_t)(n0 + 48 + col) * DIM_;

    v8f c00 = {}, c01 = {}, c02 = {}, c03 = {};
    v8f c10 = {}, c11 = {}, c12 = {}, c13 = {};
#pragma unroll 2
    for (int kk = 0; kk < DIM_; kk += 32) {
        __builtin_prefetch(ap0 + kk + 256, 0, 1);
        // All fragments live before any WMMA -> loads clause & pipeline.
        v16h a0 = cat16(ld8(ap0 + kk + hi * 16), ld8(ap0 + kk + hi * 16 + 8));
        v16h a1 = cat16(ld8(ap1 + kk + hi * 16), ld8(ap1 + kk + hi * 16 + 8));
        v16h b0 = cat16(ld8(bp0 + kk + hi * 16), ld8(bp0 + kk + hi * 16 + 8));
        v16h b1 = cat16(ld8(bp1 + kk + hi * 16), ld8(bp1 + kk + hi * 16 + 8));
        v16h b2 = cat16(ld8(bp2 + kk + hi * 16), ld8(bp2 + kk + hi * 16 + 8));
        v16h b3 = cat16(ld8(bp3 + kk + hi * 16), ld8(bp3 + kk + hi * 16 + 8));
        c00 = wmma_f16(a0, b0, c00);
        c01 = wmma_f16(a0, b1, c01);
        c02 = wmma_f16(a0, b2, c02);
        c03 = wmma_f16(a0, b3, c03);
        c10 = wmma_f16(a1, b0, c10);
        c11 = wmma_f16(a1, b1, c11);
        c12 = wmma_f16(a1, b2, c12);
        c13 = wmma_f16(a1, b3, c13);
    }

#pragma unroll
    for (int i = 0; i < 2; ++i) {
#pragma unroll
        for (int j = 0; j < 4; ++j) {
            const v8f c = (i == 0)
                ? ((j == 0) ? c00 : (j == 1) ? c01 : (j == 2) ? c02 : c03)
                : ((j == 0) ? c10 : (j == 1) ? c11 : (j == 2) ? c12 : c13);
            const int n = n0 + j * 16 + col;
            const float bn = bias[n];
            const int hsel = n >> 6;
            const int hd   = n & 63;
#pragma unroll
            for (int r = 0; r < 8; ++r) {
                int m  = m0 + i * 16 + r + hi * 8;  // C/D row mapping
                int bb = m >> 11;                   // / S_
                int s  = m & (S_ - 1);
                float v = c[r] + bn;
                if (which == 2) {
                    vh[(((size_t)bb * H_ + hsel) * HD_ + hd) * S_ + s] = (_Float16)v;
                } else {
                    _Float16* dst = (which == 0) ? qh : kh;
                    dst[(((size_t)bb * H_ + hsel) * S_ + s) * HD_ + hd] = (_Float16)v;
                }
            }
        }
    }
}

// ---------------------------------------------------------------------------
// RoPE on q and k (pairs (d, d+32)); folds 1/sqrt(HD) into q.
// ---------------------------------------------------------------------------
__global__ void rope_kernel(_Float16* __restrict__ qh,
                            _Float16* __restrict__ kh,
                            const int* __restrict__ pos) {
    int idx = blockIdx.x * blockDim.x + threadIdx.x;  // < 2*B*H*S*32
    if (idx >= 2 * B_ * H_ * S_ * 32) return;
    int j = idx & 31;         int t = idx >> 5;
    int s = t & (S_ - 1);     t >>= 11;
    int h = t & (H_ - 1);     t >>= 4;
    int b = t & (B_ - 1);     t >>= 1;
    int mat = t;              // 0 = q, 1 = k

    _Float16* base = (mat ? kh : qh) + (((size_t)b * H_ + h) * S_ + s) * HD_;
    float x0 = (float)base[j];
    float y0 = (float)base[j + 32];
    // freqs = 10000^(-j/32) = exp(-j * ln(10000)/32)
    float freq = __expf(-(float)j * (9.210340371976184f / 32.0f));
    float th   = (float)pos[b * S_ + s] * freq;
    float sn, cs;
    __sincosf(th, &sn, &cs);
    float nx = x0 * cs - y0 * sn;
    float ny = x0 * sn + y0 * cs;
    if (mat == 0) { nx *= 0.125f; ny *= 0.125f; }  // 1/sqrt(64)
    base[j]      = (_Float16)nx;
    base[j + 32] = (_Float16)ny;
}

// ---------------------------------------------------------------------------
// Flash attention: one wave per (16 query rows, head, batch).
// K/V tiles streamed into LDS by the Tensor Data Mover, double-buffered,
// synchronized with s_wait_tensorcnt. Never materializes the SxS scores.
// ---------------------------------------------------------------------------
__global__ __launch_bounds__(32)
void flash_attn_kernel(const _Float16* __restrict__ qh,
                       const _Float16* __restrict__ kh,
                       const _Float16* __restrict__ vh,
                       const float* __restrict__ mask,
                       _Float16* __restrict__ oh) {
    __shared__ __align__(16) _Float16 kls[2][32 * 64];  // K tile [s][d], 4KB each
    __shared__ __align__(16) _Float16 vls[2][64 * 32];  // V tile [d][s], 4KB each
    __shared__ __align__(16) _Float16 pls[16 * 32];     // P bounce (C-frag->A-frag)

    const int lane = threadIdx.x & 31;
    const int col  = lane & 15;
    const int hi   = lane >> 4;
    const int qtile = blockIdx.x;   // 0..127
    const int h     = blockIdx.y;
    const int b     = blockIdx.z;

    const _Float16* qbase = qh + (((size_t)b * H_ + h) * S_) * HD_;
    const _Float16* kbase = kh + (((size_t)b * H_ + h) * S_) * HD_;
    const _Float16* vbase = vh + (((size_t)b * H_ + h) * HD_) * S_;
    const float*    mrow  = mask + (size_t)b * S_;   // mask[b,0,0,:]

    // Q A-fragments over d (two 32-wide K chunks), kept in registers
    const _Float16* qrow = qbase + (size_t)(qtile * 16 + col) * HD_;
    v16h aq0 = cat16(ld8(qrow + hi * 16),      ld8(qrow + hi * 16 + 8));
    v16h aq1 = cat16(ld8(qrow + 32 + hi * 16), ld8(qrow + 32 + hi * 16 + 8));

    // TDM prologue: tiles for kc=0 (buf0) and kc=32 (buf1)
    const unsigned klo0 = lds_offset(&kls[0][0]), klo1 = lds_offset(&kls[1][0]);
    const unsigned vlo0 = lds_offset(&vls[0][0]), vlo1 = lds_offset(&vls[1][0]);
    tdm_load_2d(klo0, kbase +  0 * HD_, 64u, 32u, 64u, 64u, 32u);
    tdm_load_2d(vlo0, vbase +  0,       32u, 64u, (unsigned)S_, 32u, 64u);
    tdm_load_2d(klo1, kbase + 32 * HD_, 64u, 32u, 64u, 64u, 32u);
    tdm_load_2d(vlo1, vbase + 32,       32u, 64u, (unsigned)S_, 32u, 64u);

    v8f oacc[4] = {{}, {}, {}, {}};
    float mi[8], li[8];
#pragma unroll
    for (int r = 0; r < 8; ++r) { mi[r] = -1e30f; li[r] = 0.0f; }

    for (int kc = 0; kc < S_; kc += 32) {
        const int buf = (kc >> 5) & 1;
        // Pair for this tile done; the next pair may still be in flight.
        if (kc + 32 < S_) __builtin_amdgcn_s_wait_tensorcnt(2);
        else              __builtin_amdgcn_s_wait_tensorcnt(0);
        const _Float16* kt = &kls[buf][0];
        const _Float16* vt = &vls[buf][0];

        // ---- scores: two 16x16 tiles (keys kc..kc+15 and kc+16..kc+31) ----
        v8f s0 = {}, s1 = {};
        {
            const _Float16* kr0 = kt + (size_t)col * HD_;          // s_loc = col
            const _Float16* kr1 = kt + (size_t)(16 + col) * HD_;   // s_loc = 16+col
            v16h b00 = cat16(ld8(kr0 + hi * 16),      ld8(kr0 + hi * 16 + 8));
            v16h b01 = cat16(ld8(kr0 + 32 + hi * 16), ld8(kr0 + 32 + hi * 16 + 8));
            v16h b10 = cat16(ld8(kr1 + hi * 16),      ld8(kr1 + hi * 16 + 8));
            v16h b11 = cat16(ld8(kr1 + 32 + hi * 16), ld8(kr1 + 32 + hi * 16 + 8));
            s0 = wmma_f16(aq0, b00, s0);
            s0 = wmma_f16(aq1, b01, s0);
            s1 = wmma_f16(aq0, b10, s1);
            s1 = wmma_f16(aq1, b11, s1);
        }
        const float mv0 = mrow[kc + col];
        const float mv1 = mrow[kc + 16 + col];

        // ---- online softmax per row (row m = r + hi*8; cols across lanes) ----
        float alpha[8];
#pragma unroll
        for (int r = 0; r < 8; ++r) {
            float a  = s0[r] + mv0;
            float c2 = s1[r] + mv1;
            float mx = fmaxf(a, c2);
            mx = fmaxf(mx, __shfl_xor(mx, 1));
            mx = fmaxf(mx, __shfl_xor(mx, 2));
            mx = fmaxf(mx, __shfl_xor(mx, 4));
            mx = fmaxf(mx, __shfl_xor(mx, 8));     // reduce within lane half
            float mnew = fmaxf(mi[r], mx);
            float p0 = __expf(a  - mnew);
            float p1 = __expf(c2 - mnew);
            float rs = p0 + p1;
            rs += __shfl_xor(rs, 1);
            rs += __shfl_xor(rs, 2);
            rs += __shfl_xor(rs, 4);
            rs += __shfl_xor(rs, 8);
            alpha[r] = __expf(mi[r] - mnew);
            li[r] = li[r] * alpha[r] + rs;
            mi[r] = mnew;
            s0[r] = p0;
            s1[r] = p1;
        }
#pragma unroll
        for (int r = 0; r < 8; ++r) {
            oacc[0][r] *= alpha[r];
            oacc[1][r] *= alpha[r];
            oacc[2][r] *= alpha[r];
            oacc[3][r] *= alpha[r];
        }

        // ---- P: C-fragment -> LDS (row-major 16x32) -> A-fragment ----
        __syncthreads();
#pragma unroll
        for (int r = 0; r < 8; ++r) {
            pls[(r + hi * 8) * 32 + col]      = (_Float16)s0[r];
            pls[(r + hi * 8) * 32 + 16 + col] = (_Float16)s1[r];
        }
        __syncthreads();
        const _Float16* prow = &pls[col * 32];
        v16h ap = cat16(ld8(prow + hi * 16), ld8(prow + hi * 16 + 8));

        // ---- O += P * V  (V tile [d][s] in LDS: contiguous along s) ----
#pragma unroll
        for (int j = 0; j < 4; ++j) {
            const _Float16* vr = vt + (size_t)(j * 16 + col) * 32 + hi * 16;
            v16h bv = cat16(ld8(vr), ld8(vr + 8));
            oacc[j] = wmma_f16(ap, bv, oacc[j]);
        }

        // ---- TDM: issue tiles for kc+64 into the buffer we just consumed ----
        if (kc + 64 < S_) {
            const unsigned klo = buf ? klo1 : klo0;
            const unsigned vlo = buf ? vlo1 : vlo0;
            tdm_load_2d(klo, kbase + (size_t)(kc + 64) * HD_, 64u, 32u, 64u, 64u, 32u);
            tdm_load_2d(vlo, vbase + (kc + 64),               32u, 64u, (unsigned)S_, 32u, 64u);
        }
    }

    // ---- normalize, store O as f16 [b*S+s][h*64+d] for the out-proj GEMM ----
#pragma unroll
    for (int r = 0; r < 8; ++r) {
        float inv = 1.0f / li[r];
        int srow = qtile * 16 + r + hi * 8;
        _Float16* dst = oh + ((size_t)b * S_ + srow) * DIM_ + h * HD_;
        dst[0 * 16 + col] = (_Float16)(oacc[0][r] * inv);
        dst[1 * 16 + col] = (_Float16)(oacc[1][r] * inv);
        dst[2 * 16 + col] = (_Float16)(oacc[2][r] * inv);
        dst[3 * 16 + col] = (_Float16)(oacc[3][r] * inv);
    }
}

// ---------------------------------------------------------------------------
// Output projection GEMM: [4096 x 1024] @ [1024 x 1024] + bias -> f32 out.
// Same 32x64-per-wave tile structure as the QKV GEMM.
// ---------------------------------------------------------------------------
__global__ __launch_bounds__(32)
void out_proj_kernel(const _Float16* __restrict__ oh,
                     const _Float16* __restrict__ wot,
                     const float* __restrict__ bo,
                     float* __restrict__ out) {
    const int lane = threadIdx.x & 31;
    const int col  = lane & 15;
    const int hi   = lane >> 4;
    const int n0 = blockIdx.x * 64;  // 16 strips
    const int m0 = blockIdx.y * 32;  // 128 strips

    const _Float16* ap0 = oh + (size_t)(m0 + col) * DIM_;
    const _Float16* ap1 = ap0 + (size_t)16 * DIM_;
    const _Float16* bp0 = wot + (size_t)(n0 +  0 + col) * DIM_;
    const _Float16* bp1 = wot + (size_t)(n0 + 16 + col) * DIM_;
    const _Float16* bp2 = wot + (size_t)(n0 + 32 + col) * DIM_;
    const _Float16* bp3 = wot + (size_t)(n0 + 48 + col) * DIM_;

    v8f c00 = {}, c01 = {}, c02 = {}, c03 = {};
    v8f c10 = {}, c11 = {}, c12 = {}, c13 = {};
#pragma unroll 2
    for (int kk = 0; kk < DIM_; kk += 32) {
        __builtin_prefetch(ap0 + kk + 256, 0, 1);
        v16h a0 = cat16(ld8(ap0 + kk + hi * 16), ld8(ap0 + kk + hi * 16 + 8));
        v16h a1 = cat16(ld8(ap1 + kk + hi * 16), ld8(ap1 + kk + hi * 16 + 8));
        v16h b0 = cat16(ld8(bp0 + kk + hi * 16), ld8(bp0 + kk + hi * 16 + 8));
        v16h b1 = cat16(ld8(bp1 + kk + hi * 16), ld8(bp1 + kk + hi * 16 + 8));
        v16h b2 = cat16(ld8(bp2 + kk + hi * 16), ld8(bp2 + kk + hi * 16 + 8));
        v16h b3 = cat16(ld8(bp3 + kk + hi * 16), ld8(bp3 + kk + hi * 16 + 8));
        c00 = wmma_f16(a0, b0, c00);
        c01 = wmma_f16(a0, b1, c01);
        c02 = wmma_f16(a0, b2, c02);
        c03 = wmma_f16(a0, b3, c03);
        c10 = wmma_f16(a1, b0, c10);
        c11 = wmma_f16(a1, b1, c11);
        c12 = wmma_f16(a1, b2, c12);
        c13 = wmma_f16(a1, b3, c13);
    }

#pragma unroll
    for (int i = 0; i < 2; ++i) {
#pragma unroll
        for (int j = 0; j < 4; ++j) {
            const v8f c = (i == 0)
                ? ((j == 0) ? c00 : (j == 1) ? c01 : (j == 2) ? c02 : c03)
                : ((j == 0) ? c10 : (j == 1) ? c11 : (j == 2) ? c12 : c13);
            const int n = n0 + j * 16 + col;
            const float bn = bo[n];
#pragma unroll
            for (int r = 0; r < 8; ++r) {
                int m = m0 + i * 16 + r + hi * 8;
                out[(size_t)m * DIM_ + n] = c[r] + bn;
            }
        }
    }
}

// ---------------------------------------------------------------------------
// Host launcher
// ---------------------------------------------------------------------------
extern "C" void kernel_launch(void* const* d_in, const int* in_sizes, int n_in,
                              void* d_out, int out_size, void* d_ws, size_t ws_size,
                              hipStream_t stream) {
    const float* x    = (const float*)d_in[0];
    const int*   pos  = (const int*)d_in[1];
    const float* mask = (const float*)d_in[2];
    const float* wq   = (const float*)d_in[3];
    const float* bq   = (const float*)d_in[4];
    const float* wk   = (const float*)d_in[5];
    const float* bk   = (const float*)d_in[6];
    const float* wv   = (const float*)d_in[7];
    const float* bv   = (const float*)d_in[8];
    const float* wo   = (const float*)d_in[9];
    const float* bo   = (const float*)d_in[10];
    float* out = (float*)d_out;

    // Workspace layout (f16 buffers, all 16B aligned)
    char* ws = (char*)d_ws;
    const size_t MB = 1024 * 1024;
    _Float16* xh  = (_Float16*)(ws + 0);          //  8 MB  x as f16 [4096][1024]
    _Float16* wqt = (_Float16*)(ws + 8  * MB);    //  2 MB  Wq^T f16 [n][k]
    _Float16* wkt = (_Float16*)(ws + 10 * MB);    //  2 MB
    _Float16* wvt = (_Float16*)(ws + 12 * MB);    //  2 MB
    _Float16* wot = (_Float16*)(ws + 14 * MB);    //  2 MB
    _Float16* qh  = (_Float16*)(ws + 16 * MB);    //  8 MB  q f16 [b][h][s][d]
    _Float16* kh  = (_Float16*)(ws + 24 * MB);    //  8 MB  k f16 [b][h][s][d]
    _Float16* vh  = (_Float16*)(ws + 32 * MB);    //  8 MB  v f16 [b][h][d][s]
    _Float16* oh  = (_Float16*)(ws + 40 * MB);    //  8 MB  attn out f16 [m][h*64+d]

    const int NX = B_ * S_ * DIM_;                // 4,194,304
    cvt_f32_to_f16_kernel<<<NX / 256, 256, 0, stream>>>(x, xh, NX);

    const int NW = DIM_ * DIM_;                   // 1,048,576
    transpose_w_kernel<<<NW / 256, 256, 0, stream>>>(wq, wqt);
    transpose_w_kernel<<<NW / 256, 256, 0, stream>>>(wk, wkt);
    transpose_w_kernel<<<NW / 256, 256, 0, stream>>>(wv, wvt);
    transpose_w_kernel<<<NW / 256, 256, 0, stream>>>(wo, wot);

    // QKV projections: 16 n-strips x 128 m-strips x 3 matrices
    qkv_gemm_kernel<<<dim3(16, 128, 3), 32, 0, stream>>>(
        xh, wqt, wkt, wvt, bq, bk, bv, qh, kh, vh);

    const int NR = 2 * B_ * H_ * S_ * 32;         // 4,194,304 rotation pairs
    rope_kernel<<<NR / 256, 256, 0, stream>>>(qh, kh, pos);

    // Flash attention: (S/16)=128 q-tiles x 16 heads x 2 batches
    flash_attn_kernel<<<dim3(S_ / 16, H_, B_), 32, 0, stream>>>(
        qh, kh, vh, mask, oh);

    // Output projection
    out_proj_kernel<<<dim3(16, 128), 32, 0, stream>>>(oh, wot, bo, out);
}
